// DiT_Attention_61203283968573
// MI455X (gfx1250) — matrix operand order repaired and checked by reading, not compiled, not run on hardware
//
#include <hip/hip_runtime.h>

// ---------------------------------------------------------------------------
// CDNA5 (gfx1250) DiT attention: bf16 WMMA pipeline
//   convert -> QKV projection GEMMs -> flash attention -> output projection
// All matrix math on V_WMMA_F32_16X16X32_BF16 (wave32).
// ---------------------------------------------------------------------------

typedef __bf16 bf16_t;
typedef __attribute__((ext_vector_type(16))) __bf16 v16bf;
typedef __attribute__((ext_vector_type(8)))  float  v8f;

union Frag16 { v16bf v; uint4 q[2]; };

static __device__ __forceinline__ unsigned short f32_bf16(float f) {
  unsigned u = __float_as_uint(f);
  u += 0x7FFFu + ((u >> 16) & 1u);           // round-to-nearest-even
  return (unsigned short)(u >> 16);
}

// A operand (16x32 bf16, MxK). ISA layout: lane holds row m=lane&15;
// lanes 0-15: K = {kb..kb+7, 16+kb..16+kb+7} with kb=0; lanes 16-31: kb=8.
static __device__ __forceinline__ v16bf load_a_frag(const unsigned short* base,
                                                    int stride, int lane) {
  const int m = lane & 15, kb = (lane >> 4) << 3;
  const unsigned short* p = base + m * stride + kb;
  Frag16 f;
  f.q[0] = *(const uint4*)(p);
  f.q[1] = *(const uint4*)(p + 16);
  return f.v;
}

// B operand (32x16 bf16, KxN) gathered from an N-major [n][k] tile:
// lane holds col n=lane&15, K = {kb..kb+7, 16+kb..16+kb+7}, kb=(lane>>4)*8.
static __device__ __forceinline__ v16bf load_bT_frag(const unsigned short* rowNk,
                                                     int stride, int lane) {
  const int n = lane & 15, kb = (lane >> 4) << 3;
  const unsigned short* p = rowNk + n * stride + kb;
  Frag16 f;
  f.q[0] = *(const uint4*)(p);
  f.q[1] = *(const uint4*)(p + 16);
  return f.v;
}

static __device__ __forceinline__ v8f wmma_bf16(v16bf a, v16bf b, v8f c) {
  return __builtin_amdgcn_wmma_f32_16x16x32_bf16(false, a, false, b,
                                                 (short)0, c, false, false);
}

// ---------------------------------------------------------------------------
// f32 -> bf16 elementwise convert
// ---------------------------------------------------------------------------
__global__ __launch_bounds__(256) void k_convert(const float* __restrict__ src,
                                                 unsigned short* __restrict__ dst,
                                                 int n) {
  const int i = (blockIdx.x * 256 + threadIdx.x) * 4;
  if (i + 3 < n) {
    float4 v = *(const float4*)(src + i);
    unsigned long long r = (unsigned long long)f32_bf16(v.x)
                         | ((unsigned long long)f32_bf16(v.y) << 16)
                         | ((unsigned long long)f32_bf16(v.z) << 32)
                         | ((unsigned long long)f32_bf16(v.w) << 48);
    *(unsigned long long*)(dst + i) = r;
  }
}

// ---------------------------------------------------------------------------
// W[k][n] f32  ->  Wt[n][k] bf16  (LDS-tiled transpose, 64x64 per block)
// ---------------------------------------------------------------------------
__global__ __launch_bounds__(256) void k_transpose_bf16(const float* __restrict__ W,
                                                        unsigned short* __restrict__ Wt,
                                                        int K, int N) {
  __shared__ unsigned short T[64 * 72];
  const int k0 = blockIdx.x * 64, n0 = blockIdx.y * 64;
  const int t = threadIdx.x;
  {
    const int r = t >> 2, cs = (t & 3) << 4;
    const float* p = W + (size_t)(k0 + r) * N + n0 + cs;
#pragma unroll
    for (int c4 = 0; c4 < 4; ++c4) {
      float4 v = *(const float4*)(p + c4 * 4);
      T[r * 72 + cs + c4 * 4 + 0] = f32_bf16(v.x);
      T[r * 72 + cs + c4 * 4 + 1] = f32_bf16(v.y);
      T[r * 72 + cs + c4 * 4 + 2] = f32_bf16(v.z);
      T[r * 72 + cs + c4 * 4 + 3] = f32_bf16(v.w);
    }
  }
  __syncthreads();
  {
    const int n = t >> 2, cs = (t & 3) << 4;
    union { unsigned short h[16]; uint4 q[2]; } tmp;
#pragma unroll
    for (int c = 0; c < 16; ++c) tmp.h[c] = T[(cs + c) * 72 + n];
    unsigned short* dp = Wt + (size_t)(n0 + n) * K + k0 + cs;
    *(uint4*)(dp)     = tmp.q[0];
    *(uint4*)(dp + 8) = tmp.q[1];
  }
}

// ---------------------------------------------------------------------------
// QKV projection: C[8192 x 1024] = XC_bf16 x Wt.  grid.z selects one of the
// 6 projections. Q/K written as [B,H,S,128] (Q pre-scaled by 1/sqrt(64));
// V written transposed as [B,H,128,S] for the attention PV operand.
// 128x64 tile, 4 waves x 32 rows (B-fragments reused by both row groups).
// ---------------------------------------------------------------------------
__global__ __launch_bounds__(128) void k_proj_qkv(const unsigned short* __restrict__ XC,
                                                  const unsigned short* __restrict__ W6T,
                                                  unsigned short* __restrict__ Qb,
                                                  unsigned short* __restrict__ Kb,
                                                  unsigned short* __restrict__ Vt) {
  __shared__ unsigned short As[128 * 40];
  __shared__ unsigned short Bs[64 * 40];
  __shared__ unsigned short Cs[128 * 72];

  const int z = blockIdx.z;                 // 0..5: Qx,Kx,Vx,Qc,Kc,Vc
  const int which = z % 3;                  // 0=Q 1=K 2=V
  const int off = (z >= 3) ? 64 : 0;        // concat offset in 128-wide head dim
  const unsigned short* A  = XC + (size_t)(z >= 3 ? 1 : 0) * (8192u * 1024u);
  const unsigned short* Bt = W6T + (size_t)z * (1024u * 1024u); // [n][k]

  const int m0 = blockIdx.x * 128;          // row block in [0,8192)
  const int n0 = blockIdx.y * 64;           // col block in [0,1024)
  const int t = threadIdx.x, lane = t & 31, wv = t >> 5;

  const v8f z8 = {0.f, 0.f, 0.f, 0.f, 0.f, 0.f, 0.f, 0.f};
  v8f acc0[4] = {z8, z8, z8, z8};
  v8f acc1[4] = {z8, z8, z8, z8};

  for (int k0 = 0; k0 < 1024; k0 += 32) {
    const int r = t >> 1, seg = (t & 1) << 4;
#pragma unroll
    for (int rr = 0; rr < 128; rr += 64) {   // A tile: 128 x 32
      const uint4* ga = (const uint4*)(A + (size_t)(m0 + rr + r) * 1024 + k0 + seg);
      uint4 a0 = ga[0], a1 = ga[1];
      *(uint4*)(As + (rr + r) * 40 + seg)     = a0;
      *(uint4*)(As + (rr + r) * 40 + seg + 8) = a1;
    }
    {                                        // B tile: 64 x 32 ([n][k])
      const uint4* gb = (const uint4*)(Bt + (size_t)(n0 + r) * 1024 + k0 + seg);
      uint4 b0 = gb[0], b1 = gb[1];
      *(uint4*)(Bs + r * 40 + seg)     = b0;
      *(uint4*)(Bs + r * 40 + seg + 8) = b1;
    }
    __syncthreads();
    v16bf af0 = load_a_frag(As + (wv * 32) * 40, 40, lane);
    v16bf af1 = load_a_frag(As + (wv * 32 + 16) * 40, 40, lane);
    v16bf bf0 = load_bT_frag(Bs + 0 * 16 * 40, 40, lane);
    v16bf bf1 = load_bT_frag(Bs + 1 * 16 * 40, 40, lane);
    v16bf bf2 = load_bT_frag(Bs + 2 * 16 * 40, 40, lane);
    v16bf bf3 = load_bT_frag(Bs + 3 * 16 * 40, 40, lane);
    acc0[0] = wmma_bf16(af0, bf0, acc0[0]);
    acc0[1] = wmma_bf16(af0, bf1, acc0[1]);
    acc0[2] = wmma_bf16(af0, bf2, acc0[2]);
    acc0[3] = wmma_bf16(af0, bf3, acc0[3]);
    acc1[0] = wmma_bf16(af1, bf0, acc1[0]);
    acc1[1] = wmma_bf16(af1, bf1, acc1[1]);
    acc1[2] = wmma_bf16(af1, bf2, acc1[2]);
    acc1[3] = wmma_bf16(af1, bf3, acc1[3]);
    __syncthreads();
  }

  const int b = m0 >> 11, s0 = m0 & 2047, h = n0 >> 6;

  if (which < 2) {
    unsigned short* D = (which == 0) ? Qb : Kb;
    const float sc = (which == 0) ? 0.125f : 1.0f;   // fold 1/sqrt(DK) into Q
#pragma unroll
    for (int rg = 0; rg < 2; ++rg) {
      const v8f* accp = rg ? acc1 : acc0;
#pragma unroll
      for (int nt = 0; nt < 4; ++nt) {
#pragma unroll
        for (int v = 0; v < 8; ++v) {
          const int row = m0 + wv * 32 + rg * 16 + ((lane >> 4) << 3) + v;
          const int s = row & 2047;
          D[((size_t)(b * 16 + h) * 2048 + s) * 128 + off + nt * 16 + (lane & 15)] =
              f32_bf16(accp[nt][v] * sc);
        }
      }
    }
  } else {
    // stage C tile, then coalesced transposed writeout to Vt[B,H,128,S]
#pragma unroll
    for (int rg = 0; rg < 2; ++rg) {
      const v8f* accp = rg ? acc1 : acc0;
#pragma unroll
      for (int nt = 0; nt < 4; ++nt)
#pragma unroll
        for (int v = 0; v < 8; ++v) {
          const int rr = wv * 32 + rg * 16 + ((lane >> 4) << 3) + v;
          Cs[rr * 72 + nt * 16 + (lane & 15)] = f32_bf16(accp[nt][v]);
        }
    }
    __syncthreads();
    const int j = t >> 1, part = t & 1;     // j = dv within tile, part = key half
    union { unsigned short h16[64]; uint4 q[8]; } pk;
#pragma unroll
    for (int r = 0; r < 64; ++r) pk.h16[r] = Cs[(part * 64 + r) * 72 + j];
    unsigned short* Dp = Vt + ((size_t)(b * 16 + h) * 128 + off + j) * 2048 + s0 + part * 64;
#pragma unroll
    for (int i = 0; i < 8; ++i) *(uint4*)(Dp + i * 8) = pk.q[i];
  }
}

// ---------------------------------------------------------------------------
// Flash attention. One block = 64 query rows of one (b,h); 4 waves x 16 rows.
// Streams 64-key tiles of K [S,128] and Vt [128,S]; online softmax; all GEMM
// work in bf16 WMMA with f32 accumulators held in registers (O: 16x128/wave).
// ---------------------------------------------------------------------------
__global__ __launch_bounds__(128) void k_attn(const unsigned short* __restrict__ Qg,
                                              const unsigned short* __restrict__ Kg,
                                              const unsigned short* __restrict__ Vtg,
                                              unsigned short* __restrict__ ATT) {
  __shared__ unsigned short Qs[64 * 136];
  __shared__ unsigned short Ks[64 * 136];
  __shared__ unsigned short Vs[128 * 72];
  __shared__ unsigned short Ps[4 * 16 * 72];

  const int t = threadIdx.x, lane = t & 31, wv = t >> 5;
  const int q0 = blockIdx.x * 64;
  const int bh = blockIdx.y;
  const size_t qkbase = (size_t)bh * 2048 * 128;
  const size_t vbase  = (size_t)bh * 128 * 2048;

  { // load Q tile once
    const int r = t >> 1, half = (t & 1) << 6;
    const uint4* g = (const uint4*)(Qg + qkbase + (size_t)(q0 + r) * 128 + half);
#pragma unroll
    for (int i = 0; i < 8; ++i) *(uint4*)(Qs + r * 136 + half + i * 8) = g[i];
  }
  __syncthreads();
  v16bf qf[4];
#pragma unroll
  for (int kc = 0; kc < 4; ++kc)
    qf[kc] = load_a_frag(Qs + wv * 16 * 136 + kc * 32, 136, lane);

  const v8f z8 = {0.f, 0.f, 0.f, 0.f, 0.f, 0.f, 0.f, 0.f};
  v8f o[8] = {z8, z8, z8, z8, z8, z8, z8, z8};
  float rmax[8], rsum[8];
#pragma unroll
  for (int v = 0; v < 8; ++v) { rmax[v] = -1.0e30f; rsum[v] = 0.f; }

  unsigned short* myP = Ps + wv * (16 * 72);

  for (int kt = 0; kt < 2048; kt += 64) {
    { // K tile [64 keys x 128] and Vt tile [128 dv x 64 keys]
      const int r = t >> 1, half = (t & 1) << 6;
      const uint4* gk = (const uint4*)(Kg + qkbase + (size_t)(kt + r) * 128 + half);
#pragma unroll
      for (int i = 0; i < 8; ++i) *(uint4*)(Ks + r * 136 + half + i * 8) = gk[i];
      const uint4* gv = (const uint4*)(Vtg + vbase + (size_t)t * 2048 + kt);
#pragma unroll
      for (int i = 0; i < 8; ++i) *(uint4*)(Vs + t * 72 + i * 8) = gv[i];
    }
    __syncthreads();

    // scores: 16q x 64key per wave. kc outer -> 4 independent WMMAs per step.
    v8f sc[4] = {z8, z8, z8, z8};
#pragma unroll
    for (int kc = 0; kc < 4; ++kc) {
      v16bf kf0 = load_bT_frag(Ks + 0 * 16 * 136 + kc * 32, 136, lane);
      v16bf kf1 = load_bT_frag(Ks + 1 * 16 * 136 + kc * 32, 136, lane);
      v16bf kf2 = load_bT_frag(Ks + 2 * 16 * 136 + kc * 32, 136, lane);
      v16bf kf3 = load_bT_frag(Ks + 3 * 16 * 136 + kc * 32, 136, lane);
      sc[0] = wmma_bf16(qf[kc], kf0, sc[0]);
      sc[1] = wmma_bf16(qf[kc], kf1, sc[1]);
      sc[2] = wmma_bf16(qf[kc], kf2, sc[2]);
      sc[3] = wmma_bf16(qf[kc], kf3, sc[3]);
    }

    // online softmax (row r lives across one 16-lane half, element v)
#pragma unroll
    for (int v = 0; v < 8; ++v) {
      float m = fmaxf(fmaxf(sc[0][v], sc[1][v]), fmaxf(sc[2][v], sc[3][v]));
      m = fmaxf(m, __shfl_xor(m, 1));
      m = fmaxf(m, __shfl_xor(m, 2));
      m = fmaxf(m, __shfl_xor(m, 4));
      m = fmaxf(m, __shfl_xor(m, 8));
      const float nm = fmaxf(rmax[v], m);
      const float alpha = __expf(rmax[v] - nm);
      rmax[v] = nm;
      float ps = 0.f;
#pragma unroll
      for (int nt = 0; nt < 4; ++nt) {
        const float p = __expf(sc[nt][v] - nm);
        sc[nt][v] = p;
        ps += p;
      }
      ps += __shfl_xor(ps, 1);
      ps += __shfl_xor(ps, 2);
      ps += __shfl_xor(ps, 4);
      ps += __shfl_xor(ps, 8);
      rsum[v] = rsum[v] * alpha + ps;
#pragma unroll
      for (int nt = 0; nt < 8; ++nt) o[nt][v] *= alpha;
    }

    // stage P (C-layout -> row-major) in this wave's private LDS region
    {
      const int rr = (lane >> 4) << 3;
#pragma unroll
      for (int nt = 0; nt < 4; ++nt)
#pragma unroll
        for (int v = 0; v < 8; ++v)
          myP[(rr + v) * 72 + nt * 16 + (lane & 15)] = f32_bf16(sc[nt][v]);
    }

    // O += P x V  (contraction over 64 keys = 2 chunks of 32);
    // batch the 8 V fragments ahead of the 8 independent WMMAs.
#pragma unroll
    for (int kc = 0; kc < 2; ++kc) {
      v16bf pf = load_a_frag(myP + kc * 32, 72, lane);
      v16bf vf[8];
#pragma unroll
      for (int nt = 0; nt < 8; ++nt)
        vf[nt] = load_bT_frag(Vs + nt * 16 * 72 + kc * 32, 72, lane);
#pragma unroll
      for (int nt = 0; nt < 8; ++nt)
        o[nt] = wmma_bf16(pf, vf[nt], o[nt]);
    }
    __syncthreads();
  }

  { // normalize and write [B,S,H*128] bf16
    const int h = bh & 15, b = bh >> 4;
#pragma unroll
    for (int nt = 0; nt < 8; ++nt)
#pragma unroll
      for (int v = 0; v < 8; ++v) {
        const int q = q0 + wv * 16 + ((lane >> 4) << 3) + v;
        ATT[(size_t)(b * 2048 + q) * 2048 + h * 128 + nt * 16 + (lane & 15)] =
            f32_bf16(o[nt][v] / rsum[v]);
      }
  }
}

// ---------------------------------------------------------------------------
// Output projection: out[8192 x 1024] = ATT[8192 x 2048] x Wproj + bias (f32)
// 128x64 tile, 4 waves x 32 rows.
// ---------------------------------------------------------------------------
__global__ __launch_bounds__(128) void k_proj_out(const unsigned short* __restrict__ ATT,
                                                  const unsigned short* __restrict__ Wpt,
                                                  const float* __restrict__ bias,
                                                  float* __restrict__ out) {
  __shared__ unsigned short As[128 * 40];
  __shared__ unsigned short Bs[64 * 40];
  const int m0 = blockIdx.x * 128, n0 = blockIdx.y * 64;
  const int t = threadIdx.x, lane = t & 31, wv = t >> 5;

  const v8f z8 = {0.f, 0.f, 0.f, 0.f, 0.f, 0.f, 0.f, 0.f};
  v8f acc0[4] = {z8, z8, z8, z8};
  v8f acc1[4] = {z8, z8, z8, z8};

  for (int k0 = 0; k0 < 2048; k0 += 32) {
    const int r = t >> 1, seg = (t & 1) << 4;
#pragma unroll
    for (int rr = 0; rr < 128; rr += 64) {
      const uint4* ga = (const uint4*)(ATT + (size_t)(m0 + rr + r) * 2048 + k0 + seg);
      uint4 a0 = ga[0], a1 = ga[1];
      *(uint4*)(As + (rr + r) * 40 + seg)     = a0;
      *(uint4*)(As + (rr + r) * 40 + seg + 8) = a1;
    }
    {
      const uint4* gb = (const uint4*)(Wpt + (size_t)(n0 + r) * 2048 + k0 + seg);
      uint4 b0 = gb[0], b1 = gb[1];
      *(uint4*)(Bs + r * 40 + seg)     = b0;
      *(uint4*)(Bs + r * 40 + seg + 8) = b1;
    }
    __syncthreads();
    v16bf af0 = load_a_frag(As + (wv * 32) * 40, 40, lane);
    v16bf af1 = load_a_frag(As + (wv * 32 + 16) * 40, 40, lane);
    v16bf bf0 = load_bT_frag(Bs + 0 * 16 * 40, 40, lane);
    v16bf bf1 = load_bT_frag(Bs + 1 * 16 * 40, 40, lane);
    v16bf bf2 = load_bT_frag(Bs + 2 * 16 * 40, 40, lane);
    v16bf bf3 = load_bT_frag(Bs + 3 * 16 * 40, 40, lane);
    acc0[0] = wmma_bf16(af0, bf0, acc0[0]);
    acc0[1] = wmma_bf16(af0, bf1, acc0[1]);
    acc0[2] = wmma_bf16(af0, bf2, acc0[2]);
    acc0[3] = wmma_bf16(af0, bf3, acc0[3]);
    acc1[0] = wmma_bf16(af1, bf0, acc1[0]);
    acc1[1] = wmma_bf16(af1, bf1, acc1[1]);
    acc1[2] = wmma_bf16(af1, bf2, acc1[2]);
    acc1[3] = wmma_bf16(af1, bf3, acc1[3]);
    __syncthreads();
  }

#pragma unroll
  for (int rg = 0; rg < 2; ++rg) {
    const v8f* accp = rg ? acc1 : acc0;
#pragma unroll
    for (int nt = 0; nt < 4; ++nt) {
      const int n = n0 + nt * 16 + (lane & 15);
      const float bv = bias[n];
#pragma unroll
      for (int v = 0; v < 8; ++v) {
        const int row = m0 + wv * 32 + rg * 16 + ((lane >> 4) << 3) + v;
        out[(size_t)row * 1024 + n] = accp[nt][v] + bv;
      }
    }
  }
}

// ---------------------------------------------------------------------------
// Workspace layout (unsigned short element offsets):
//   XC   [2][8192][1024]        @ 0
//   W6T  6 x [1024][1024]       @ 16777216
//   WPT  [1024][2048]           @ 23068672
//   Q    [B,H,S,128]            @ 25165824
//   K    [B,H,S,128]            @ 41943040
//   Vt   [B,H,128,S]            @ 58720256
//   ATT  [B,S,2048]             @ 75497472     (end: ~176 MB)
// ---------------------------------------------------------------------------
extern "C" void kernel_launch(void* const* d_in, const int* in_sizes, int n_in,
                              void* d_out, int out_size, void* d_ws, size_t ws_size,
                              hipStream_t stream) {
  (void)in_sizes; (void)n_in; (void)out_size; (void)ws_size;
  const float* x     = (const float*)d_in[0];
  const float* c     = (const float*)d_in[1];
  const float* Wproj = (const float*)d_in[8];
  const float* bproj = (const float*)d_in[9];

  unsigned short* ws  = (unsigned short*)d_ws;
  unsigned short* XC  = ws;
  unsigned short* W6T = ws + 16777216u;
  unsigned short* WPT = ws + 23068672u;
  unsigned short* Qb  = ws + 25165824u;
  unsigned short* Kb  = ws + 41943040u;
  unsigned short* Vt  = ws + 58720256u;
  unsigned short* ATT = ws + 75497472u;

  // 1) activations -> bf16
  k_convert<<<8192, 256, 0, stream>>>(x, XC, 8388608);
  k_convert<<<8192, 256, 0, stream>>>(c, XC + 8388608u, 8388608);

  // 2) weights -> transposed bf16 ([n][k])
  for (int i = 0; i < 6; ++i)
    k_transpose_bf16<<<dim3(16, 16), 256, 0, stream>>>(
        (const float*)d_in[2 + i], W6T + (size_t)i * 1048576u, 1024, 1024);
  k_transpose_bf16<<<dim3(32, 16), 256, 0, stream>>>(Wproj, WPT, 2048, 1024);

  // 3) fused QKV projections (6 GEMMs via grid.z)
  k_proj_qkv<<<dim3(64, 16, 6), 128, 0, stream>>>(XC, W6T, Qb, Kb, Vt);

  // 4) flash attention over (q-tile, b*h)
  k_attn<<<dim3(32, 64), 128, 0, stream>>>(Qb, Kb, Vt, ATT);

  // 5) output projection + bias -> f32 d_out
  k_proj_out<<<dim3(64, 16), 128, 0, stream>>>(ATT, WPT, bproj, (float*)d_out);
}
